// CausalSelfAttention_10840497455452
// MI455X (gfx1250) — matrix operand order repaired and checked by reading, not compiled, not run on hardware
//
#include <hip/hip_runtime.h>
#include <hip/hip_bf16.h>

#ifndef __has_builtin
#define __has_builtin(x) 0
#endif

#if defined(__HIP_DEVICE_COMPILE__) && __has_builtin(__builtin_amdgcn_global_load_async_to_lds_b128)
#define HAS_ASYNC_LDS 1
#else
#define HAS_ASYNC_LDS 0
#endif

// ---------------------------------------------------------------------------
// CDNA5 (gfx1250) wave32 WMMA types & helpers
// ---------------------------------------------------------------------------
typedef __bf16 v16bf __attribute__((ext_vector_type(16)));
typedef float  v8f   __attribute__((ext_vector_type(8)));
typedef int    v4i   __attribute__((ext_vector_type(4)));

union BF16Frag { v16bf v; unsigned short u[16]; };

__device__ __forceinline__ unsigned short f2bf(float f) {
  unsigned int x = __float_as_uint(f);
  unsigned int r = (x + 0x7FFFu + ((x >> 16) & 1u)) >> 16;  // round-to-nearest-even
  return (unsigned short)r;
}

__device__ __forceinline__ v8f wmma_bf16(v16bf a, v16bf b, v8f c) {
  // D = A(16x32 bf16) * B(32x16 bf16) + C(16x16 f32)
  return __builtin_amdgcn_wmma_f32_16x16x32_bf16(false, a, false, b,
                                                 (short)0, c, false, false);
}

// 16-byte global -> LDS copy, async (ASYNCcnt) when available.
// Builtin signature (from toolchain diagnostic): (v4i AS(1)*, v4i AS(3)*, Ii, Ii).
// We form the AS pointers via integer casts: LDS flat addresses carry the LDS
// byte offset in addr[31:0] (ISA aperture mapping), global flat VA == AS(1) VA.
__device__ __forceinline__ void async_ld_b128(void* lds, const void* g) {
#if HAS_ASYNC_LDS
  __builtin_amdgcn_global_load_async_to_lds_b128(
      (v4i __attribute__((address_space(1))) *)(unsigned long long)(g),
      (v4i __attribute__((address_space(3))) *)(unsigned int)(unsigned long long)(lds),
      0, 0);
#else
  *(uint4*)lds = *(const uint4*)g;
#endif
}

__device__ __forceinline__ void async_wait() {
#if HAS_ASYNC_LDS
#if __has_builtin(__builtin_amdgcn_s_wait_asynccnt)
  __builtin_amdgcn_s_wait_asynccnt(0);
#else
  asm volatile("s_wait_asynccnt 0" ::: "memory");
#endif
#endif
}

// ---------------------------------------------------------------------------
// Kernel 0: elementwise f32 -> bf16 (n multiple of 1024)
// ---------------------------------------------------------------------------
__global__ __launch_bounds__(256) void cvt_bf16_kernel(
    const float* __restrict__ in, unsigned short* __restrict__ out) {
  long i = ((long)blockIdx.x * 256 + threadIdx.x) * 4;
  float4 v = *(const float4*)(in + i);
  ushort4 r;
  r.x = f2bf(v.x); r.y = f2bf(v.y); r.z = f2bf(v.z); r.w = f2bf(v.w);
  *(ushort4*)(out + i) = r;
}

// ---------------------------------------------------------------------------
// Kernel 1: Y[M,N] = X[M,K] @ W[N,K]^T   (bf16 in, f32 accum, f32/bf16 out)
// Block tile 64(M) x 128(N), K-step 64.  8 waves, each wave 32x32 (2x2 WMMA
// per 32-k substep => 8 WMMA per stage).  Tiles staged via async-to-LDS.
// A frag (16-bit A 16x32): lane m=l%16 row; g=l/16; elem e -> K = g*16 + e
// B frag (16-bit B 32x16): lane n=l%16 col; elem e -> K = g*16 + e
// C/D (f32 16x16): lane col N=l%16, VGPR v -> row M = 8*(l/16)+v
// ---------------------------------------------------------------------------
template <bool OUT_BF16>
__global__ __launch_bounds__(256) void gemm_xwT_kernel(
    const unsigned short* __restrict__ Xb, const unsigned short* __restrict__ Wb,
    void* __restrict__ Yv, int M, int N, int K) {
  __shared__ __align__(16) unsigned short lA[64 * 64];   //  8 KB
  __shared__ __align__(16) unsigned short lB[128 * 64];  // 16 KB

  const int tid  = threadIdx.x;
  const int wv   = tid >> 5;
  const int lane = tid & 31;
  const int g    = lane >> 4;
  const int hl   = lane & 15;
  const int wvM  = wv >> 2;   // 0..1
  const int wvN  = wv & 3;    // 0..3
  const long rowBase = (long)blockIdx.y * 64;
  const long colBase = (long)blockIdx.x * 128;

  v8f acc[2][2];
  #pragma unroll
  for (int mi = 0; mi < 2; mi++)
    #pragma unroll
    for (int ni = 0; ni < 2; ni++)
      #pragma unroll
      for (int e = 0; e < 8; e++) acc[mi][ni][e] = 0.0f;

  for (int k0 = 0; k0 < K; k0 += 64) {
    // stage A (64x64) : 512 x 16B chunks, 2 per thread
    #pragma unroll
    for (int t2 = 0; t2 < 2; t2++) {
      int j = tid + t2 * 256;
      int r = j >> 3, c = (j & 7) * 8;
      async_ld_b128(&lA[r * 64 + c], Xb + (rowBase + r) * K + k0 + c);
    }
    // stage B (128x64): 1024 x 16B chunks, 4 per thread
    #pragma unroll
    for (int t2 = 0; t2 < 4; t2++) {
      int j = tid + t2 * 256;
      int r = j >> 3, c = (j & 7) * 8;
      async_ld_b128(&lB[r * 64 + c], Wb + (colBase + r) * K + k0 + c);
    }
    async_wait();
    __syncthreads();

    #pragma unroll
    for (int ks = 0; ks < 2; ks++) {
      BF16Frag a[2], bf[2];
      #pragma unroll
      for (int mi = 0; mi < 2; mi++) {
        const unsigned short* ab = &lA[(wvM * 32 + mi * 16 + hl) * 64 + ks * 32];
        #pragma unroll
        for (int e = 0; e < 8; e++) {
          a[mi].u[e]     = ab[g * 16 + e];
          a[mi].u[8 + e] = ab[g * 16 + 8 + e];
        }
      }
      #pragma unroll
      for (int ni = 0; ni < 2; ni++) {
        const unsigned short* bb =
            &lB[(wvN * 32 + ni * 16 + hl) * 64 + ks * 32 + g * 16];
        #pragma unroll
        for (int e = 0; e < 16; e++) bf[ni].u[e] = bb[e];
      }
      #pragma unroll
      for (int mi = 0; mi < 2; mi++)
        #pragma unroll
        for (int ni = 0; ni < 2; ni++)
          acc[mi][ni] = wmma_bf16(a[mi].v, bf[ni].v, acc[mi][ni]);
    }
    __syncthreads();
  }

  #pragma unroll
  for (int mi = 0; mi < 2; mi++)
    #pragma unroll
    for (int ni = 0; ni < 2; ni++)
      #pragma unroll
      for (int v = 0; v < 8; v++) {
        long row = rowBase + wvM * 32 + mi * 16 + g * 8 + v;
        long col = colBase + wvN * 32 + ni * 16 + hl;
        if (OUT_BF16)
          ((unsigned short*)Yv)[row * N + col] = f2bf(acc[mi][ni][v]);
        else
          ((float*)Yv)[row * N + col] = acc[mi][ni][v];
      }
}

// ---------------------------------------------------------------------------
// Kernel 2: per-(token,head) RMS-norm + RoPE (+optional gain), f32 -> bf16.
// One wave per item; lane i owns the RoPE pair (i, i+32) of the 64-elem head.
// ---------------------------------------------------------------------------
__global__ __launch_bounds__(256) void rmsrope_kernel(
    const float* __restrict__ in, unsigned short* __restrict__ outb,
    const float* __restrict__ gain, int nheads) {
  const int lane  = threadIdx.x & 31;
  const int item  = blockIdx.x * 8 + (threadIdx.x >> 5);
  const int token = item / nheads;
  const int h     = item % nheads;
  const int s     = token & 2047;  // position (token = b*2048+s)
  const int stride = nheads * 64;

  const float* p = in + (long)token * stride + h * 64;
  float t1 = p[lane], t2 = p[lane + 32];
  float ss = t1 * t1 + t2 * t2;
  #pragma unroll
  for (int m = 16; m >= 1; m >>= 1) ss += __shfl_xor(ss, m, 32);
  float inv = rsqrtf(ss * (1.0f / 64.0f) + 1.1920929e-07f);
  t1 *= inv; t2 *= inv;

  float invf = __powf(10000.0f, -(float)lane * (1.0f / 32.0f));
  float ang  = (float)s * invf;
  float sn, cs;
  __sincosf(ang, &sn, &cs);
  float gh = gain ? gain[h] : 1.0f;
  float o1 = (t1 * cs + t2 * sn) * gh;
  float o2 = (-t1 * sn + t2 * cs) * gh;

  unsigned short* q = outb + (long)token * stride + h * 64;
  q[lane]      = f2bf(o1);
  q[lane + 32] = f2bf(o2);
}

// ---------------------------------------------------------------------------
// Kernel 3: causal GQA flash attention (flash-attention-2 style).
// One wave owns a 16-row query tile; 8 waves/block cover 128 rows of (b,h).
// kv step = 32 keys: S-tile via 2x(2 chained) bf16 WMMAs, online softmax in
// f32, P re-laid-out C->A through per-wave LDS, PV via 4 WMMAs (K=32).
// V tile staged per-wave via async-to-LDS; next tile prefetched.
// ---------------------------------------------------------------------------
__global__ __launch_bounds__(256) void flash_attn_kernel(
    const unsigned short* __restrict__ Qb, const unsigned short* __restrict__ Kb,
    const unsigned short* __restrict__ Vb, unsigned short* __restrict__ Yb) {
  constexpr int S = 2048, HD = 64, KD = 256, D = 1024;
  __shared__ __align__(16) unsigned short lV[8][32 * 64];  // 32 KB
  __shared__ __align__(16) unsigned short lP[8][16 * 32];  //  8 KB

  const int b    = blockIdx.z;
  const int h    = blockIdx.y;
  const int kvh  = h >> 2;                 // H/KV = 4
  const int wv   = threadIdx.x >> 5;
  const int lane = threadIdx.x & 31;
  const int g    = lane >> 4;
  const int hl   = lane & 15;
  const int qbase = blockIdx.x * 128 + wv * 16;

  // Q A-fragments for both HD k-steps, held in registers for the whole loop
  BF16Frag aq[2];
  {
    const unsigned short* qrow = Qb + (long)(b * S + qbase + hl) * D + h * HD;
    #pragma unroll
    for (int ks = 0; ks < 2; ks++)
      #pragma unroll
      for (int e = 0; e < 8; e++) {
        aq[ks].u[e]     = qrow[ks * 32 + g * 16 + e];
        aq[ks].u[8 + e] = qrow[ks * 32 + g * 16 + 8 + e];
      }
  }

  float mrow[8], lrow[8];
  v8f o[4];
  #pragma unroll
  for (int v = 0; v < 8; v++) { mrow[v] = -1e30f; lrow[v] = 0.0f; }
  #pragma unroll
  for (int j = 0; j < 4; j++)
    #pragma unroll
    for (int e = 0; e < 8; e++) o[j][e] = 0.0f;

  const int nsteps = (qbase + 15) / 32 + 1;  // causal bound for this wave
  for (int t = 0; t < nsteps; ++t) {
    const int kb = t * 32;

    // stage V tile (32 keys x 64 bf16) into this wave's LDS region (row/lane)
    {
      const unsigned short* vsrc = Vb + (long)(b * S + kb + lane) * KD + kvh * HD;
      unsigned short* vdst = &lV[wv][lane * 64];
      #pragma unroll
      for (int i = 0; i < 8; i++) async_ld_b128(vdst + i * 8, vsrc + i * 8);
    }
    // prefetch next K/V tile rows while this one computes
    if (t + 1 < nsteps) {
      __builtin_prefetch(Kb + (long)(b * S + kb + 32 + lane) * KD + kvh * HD, 0, 1);
      __builtin_prefetch(Vb + (long)(b * S + kb + 32 + lane) * KD + kvh * HD, 0, 1);
    }

    // scores S = Q * K^T  (two 16x16 n-tiles, 2 chained WMMAs over HD)
    v8f sc[2];
    #pragma unroll
    for (int ni = 0; ni < 2; ni++) {
      const unsigned short* krow =
          Kb + (long)(b * S + kb + ni * 16 + hl) * KD + kvh * HD;
      BF16Frag bk0, bk1;
      #pragma unroll
      for (int e = 0; e < 16; e++) {
        bk0.u[e] = krow[g * 16 + e];       // d = g*16+e
        bk1.u[e] = krow[32 + g * 16 + e];  // d = 32+g*16+e
      }
      v8f z;
      #pragma unroll
      for (int e = 0; e < 8; e++) z[e] = 0.0f;
      sc[ni] = wmma_bf16(aq[0].v, bk0.v, z);
      sc[ni] = wmma_bf16(aq[1].v, bk1.v, sc[ni]);
    }

    // online softmax (row = qbase + 8*g + v; 16 cols live in the half-wave)
    #pragma unroll
    for (int v = 0; v < 8; v++) {
      const int qr = qbase + g * 8 + v;
      float x0 = sc[0][v] * 0.125f;  // 1/sqrt(64)
      float x1 = sc[1][v] * 0.125f;
      if (kb + hl      > qr) x0 = -1e30f;
      if (kb + 16 + hl > qr) x1 = -1e30f;
      float mx = fmaxf(x0, x1);
      #pragma unroll
      for (int d2 = 1; d2 < 16; d2 <<= 1) mx = fmaxf(mx, __shfl_xor(mx, d2, 32));
      float mn = fmaxf(mrow[v], mx);
      float p0 = __expf(x0 - mn);
      float p1 = __expf(x1 - mn);
      float rs = p0 + p1;
      #pragma unroll
      for (int d2 = 1; d2 < 16; d2 <<= 1) rs += __shfl_xor(rs, d2, 32);
      float alpha = __expf(mrow[v] - mn);
      mrow[v] = mn;
      lrow[v] = lrow[v] * alpha + rs;
      #pragma unroll
      for (int j = 0; j < 4; j++) o[j][v] *= alpha;
      lP[wv][(g * 8 + v) * 32 + hl]      = f2bf(p0);
      lP[wv][(g * 8 + v) * 32 + 16 + hl] = f2bf(p1);
    }

    // same-wave LDS ops are in-order; fence compiler + wait P stores / V async
    asm volatile("s_wait_dscnt 0" ::: "memory");
    async_wait();

    // P (16x32) re-read in A layout from LDS
    BF16Frag pa;
    {
      const unsigned short* pr = &lP[wv][hl * 32];
      #pragma unroll
      for (int e = 0; e < 8; e++) {
        pa.u[e]     = pr[g * 16 + e];
        pa.u[8 + e] = pr[g * 16 + 8 + e];
      }
    }
    // O += P * V  (4 n-tiles of HD; K=32 keys = exactly one WMMA step)
    #pragma unroll
    for (int j = 0; j < 4; j++) {
      BF16Frag bv;
      #pragma unroll
      for (int e = 0; e < 16; e++)
        bv.u[e] = lV[wv][(g * 16 + e) * 64 + j * 16 + hl];
      o[j] = wmma_bf16(pa.v, bv.v, o[j]);
    }
  }

  // epilogue: O / l  -> y (tokens x D, bf16 for the output-projection GEMM)
  #pragma unroll
  for (int j = 0; j < 4; j++)
    #pragma unroll
    for (int v = 0; v < 8; v++) {
      const int qr = qbase + g * 8 + v;
      Yb[(long)(b * S + qr) * D + h * HD + j * 16 + hl] =
          f2bf(o[j][v] / lrow[v]);
    }
}

// ---------------------------------------------------------------------------
// Launch: cvt -> QKV GEMMs -> RMS+RoPE -> flash attention -> out projection
// Workspace layout (53 MB total), all offsets 16B-aligned:
//   [ 0, 8M) x bf16        [ 8M,10M) Wq bf16   [10M,10.5M) Wk bf16
//   [10.5M,11M) Wv bf16    [11M,13M) Wp bf16   [13M,29M) q f32
//   [29M,33M) k f32        [33M,35M) v bf16    [35M,43M) q bf16
//   [43M,45M) k bf16       [45M,53M) y_attn bf16
// ---------------------------------------------------------------------------
extern "C" void kernel_launch(void* const* d_in, const int* in_sizes, int n_in,
                              void* d_out, int out_size, void* d_ws, size_t ws_size,
                              hipStream_t stream) {
  (void)in_sizes; (void)n_in; (void)out_size; (void)ws_size;
  const float* x     = (const float*)d_in[0];
  const float* Wq    = (const float*)d_in[1];
  const float* Wk    = (const float*)d_in[2];
  const float* Wv    = (const float*)d_in[3];
  const float* Wproj = (const float*)d_in[4];
  const float* qgain = (const float*)d_in[5];

  const int M = 4096;  // B * S tokens
  char* ws = (char*)d_ws;
  const unsigned long long MB = 1ull << 20;
  unsigned short* xb  = (unsigned short*)(ws);
  unsigned short* wqb = (unsigned short*)(ws + 8 * MB);
  unsigned short* wkb = (unsigned short*)(ws + 10 * MB);
  unsigned short* wvb = (unsigned short*)(ws + 10 * MB + 512 * 1024);
  unsigned short* wpb = (unsigned short*)(ws + 11 * MB);
  float*          qf  = (float*)(ws + 13 * MB);
  float*          kf  = (float*)(ws + 29 * MB);
  unsigned short* vbb = (unsigned short*)(ws + 33 * MB);
  unsigned short* qbb = (unsigned short*)(ws + 35 * MB);
  unsigned short* kbb = (unsigned short*)(ws + 43 * MB);
  unsigned short* yab = (unsigned short*)(ws + 45 * MB);

  dim3 blk(256);
  // f32 -> bf16 conversions (x and weights), 1024 elems per block
  cvt_bf16_kernel<<<dim3(4096 * 1024 / 1024), blk, 0, stream>>>(x, xb);
  cvt_bf16_kernel<<<dim3(1024 * 1024 / 1024), blk, 0, stream>>>(Wq, wqb);
  cvt_bf16_kernel<<<dim3(256 * 1024 / 1024),  blk, 0, stream>>>(Wk, wkb);
  cvt_bf16_kernel<<<dim3(256 * 1024 / 1024),  blk, 0, stream>>>(Wv, wvb);
  cvt_bf16_kernel<<<dim3(1024 * 1024 / 1024), blk, 0, stream>>>(Wproj, wpb);

  gemm_xwT_kernel<false><<<dim3(1024 / 128, M / 64), blk, 0, stream>>>(xb, wqb, qf, M, 1024, 1024);
  gemm_xwT_kernel<false><<<dim3(256 / 128,  M / 64), blk, 0, stream>>>(xb, wkb, kf, M, 256, 1024);
  gemm_xwT_kernel<true ><<<dim3(256 / 128,  M / 64), blk, 0, stream>>>(xb, wvb, vbb, M, 256, 1024);

  rmsrope_kernel<<<dim3(M * 16 / 8), blk, 0, stream>>>(qf, qbb, qgain, 16);
  rmsrope_kernel<<<dim3(M * 4 / 8),  blk, 0, stream>>>(kf, kbb, nullptr, 4);

  flash_attn_kernel<<<dim3(16, 16, 2), blk, 0, stream>>>(qbb, kbb, vbb, yab);

  gemm_xwT_kernel<false><<<dim3(1024 / 128, M / 64), blk, 0, stream>>>(yab, wpb, (float*)d_out, M, 1024, 1024);
}
